// glob_feat_77464030151167
// MI455X (gfx1250) — matrix operand order repaired and checked by reading, not compiled, not run on hardware
//
#include <hip/hip_runtime.h>

#define DIMC 128
#define KOFF 27
#define NBMAX 8

typedef __bf16 bf16_t;
typedef __attribute__((ext_vector_type(16))) __bf16 v16bf;
typedef __attribute__((ext_vector_type(8)))  float  v8f;
typedef __attribute__((ext_vector_type(4)))  float  f32x4;
typedef __attribute__((ext_vector_type(4)))  int    v4i;

#define AS1 __attribute__((address_space(1)))
#define AS3 __attribute__((address_space(3)))

#if defined(__has_builtin)
#if __has_builtin(__builtin_amdgcn_global_load_async_to_lds_b128)
#define HAVE_ASYNC_LDS 1
#endif
#if __has_builtin(__builtin_amdgcn_s_wait_asynccnt)
#define HAVE_WAIT_ASYNC 1
#endif
#endif

union AFrag { v16bf v; f32x4 q[2]; };

// A fragment (16x32 bf16, contiguous k order, same as B packing): lanes 0-15 hold row m=lane,
// K = c*32 + {0..15}; lanes 16-31 hold K = c*32 + {16..31}.
__device__ __forceinline__ v16bf load_a_frag(const bf16_t* __restrict__ row,
                                             int cch, int lane) {
  const f32x4* p = (const f32x4*)(row + cch * 32 + ((lane >> 4) << 4));
  AFrag a; a.q[0] = p[0]; a.q[1] = p[1];
  return a.v;
}

// B fragment from LDS-staged packed weights: contiguous 32B per lane (ds_load_b128 x2).
__device__ __forceinline__ v16bf load_b_lds(const bf16_t* s, int cch, int t, int lane) {
  const f32x4* p = (const f32x4*)(s + (((cch << 3) + t) << 9) + (lane << 4));
  AFrag b; b.q[0] = p[0]; b.q[1] = p[1];
  return b.v;
}

// 16B granule copy global -> LDS, async where the toolchain exposes it.
// Builtin signature (from the toolchain diagnostic): (v4i AS1*, v4i AS3*, int, int).
__device__ __forceinline__ void stage16(const f32x4* __restrict__ src, f32x4* dst) {
#if defined(HAVE_ASYNC_LDS)
  __builtin_amdgcn_global_load_async_to_lds_b128((AS1 v4i*)src, (AS3 v4i*)dst, 0, 0);
#else
  *dst = *src;
#endif
}

__device__ __forceinline__ void stage_fence() {
#if defined(HAVE_WAIT_ASYNC)
  __builtin_amdgcn_s_wait_asynccnt(0);
#endif
  __syncthreads();
}

// ---------------- prep kernels ----------------

__global__ __launch_bounds__(256) void cast_bf16_kernel(const float* __restrict__ x,
                                                        bf16_t* __restrict__ y, long n) {
  long i = (long)blockIdx.x * 256 + threadIdx.x;
  if (i < n) y[i] = (bf16_t)x[i];
}

// Pack W[k][cin][cout] (f32, row-major) into WMMA-B lane layout (bf16):
// P[k][cch][t][lane][e]; lane<16: col=t*16+lane, kk=cch*32+e ; lane>=16: kk += 16.
__global__ __launch_bounds__(256) void pack_w_kernel(const float* __restrict__ W,
                                                     bf16_t* __restrict__ P, int numK) {
  long i = (long)blockIdx.x * 256 + threadIdx.x;
  long total = (long)numK * 4 * 8 * 512;
  if (i >= total) return;
  int e    = (int)(i & 15);
  int lane = (int)((i >> 4) & 31);
  int t    = (int)((i >> 9) & 7);
  int c    = (int)((i >> 12) & 3);
  int k    = (int)(i >> 14);
  int col = t * 16 + (lane & 15);
  int kk  = c * 32 + ((lane >> 4) << 4) + e;
  P[i] = (bf16_t)W[((long)k * DIMC + kk) * DIMC + col];
}

// ---------------- dual sparse conv: gather -> 2x WMMA GEMM -> scatter-add ----------------
// grid = (rowTileBlocks, KOFF): one k per block -> stage both weight matrices in LDS.

__global__ __launch_bounds__(256)
void spconv_dual_kernel(const bf16_t* __restrict__ featsb,
                        const bf16_t* __restrict__ pW1,
                        const bf16_t* __restrict__ pW2,
                        const int* __restrict__ in_idx,
                        const int* __restrict__ out_idx,
                        float* __restrict__ acc1,
                        float* __restrict__ acc2,
                        int L, int tilesPerK) {
  __shared__ bf16_t sW[2 * 16384];       // 64 KB: [0]=W1[k], [1]=W2[k]
  const int k = blockIdx.y;

  {
    const f32x4* g1 = (const f32x4*)(pW1 + (long)k * 16384);
    const f32x4* g2 = (const f32x4*)(pW2 + (long)k * 16384);
    f32x4* s = (f32x4*)sW;
    for (int i = threadIdx.x; i < 2048; i += 256)
      stage16((i < 1024 ? g1 + i : g2 + (i - 1024)), s + i);
  }
  stage_fence();

  int wave = threadIdx.x >> 5;
  int lane = threadIdx.x & 31;
  int tile = blockIdx.x * 8 + wave;
  if (tile >= tilesPerK) return;         // after the barrier: safe
  int base = tile * 16;

  int r = base + (lane & 15);
  int ii = (r < L) ? in_idx[(long)k * L + r] : 0;   // padded pairs gather row 0
  const bf16_t* arow = featsb + (long)ii * DIMC;
  if (r + 16 < L) __builtin_prefetch(&in_idx[(long)k * L + r + 16], 0, 0);

  const bf16_t* w1s = sW;
  const bf16_t* w2s = sW + 16384;

  v8f c1[8] = {};
  v8f c2[8] = {};
#pragma unroll
  for (int cch = 0; cch < 4; ++cch) {
    v16bf a = load_a_frag(arow, cch, lane);          // one gather, two GEMMs
#pragma unroll
    for (int t = 0; t < 8; ++t) {
      v16bf bf1 = load_b_lds(w1s, cch, t, lane);
      c1[t] = __builtin_amdgcn_wmma_f32_16x16x32_bf16(false, a, false, bf1,
                                                      (short)0, c1[t], false, false);
      v16bf bf2 = load_b_lds(w2s, cch, t, lane);
      c2[t] = __builtin_amdgcn_wmma_f32_16x16x32_bf16(false, a, false, bf2,
                                                      (short)0, c2[t], false, false);
    }
  }

  // C layout: vgpr j, lane l -> row = j + 8*(l>>4), col = t*16 + (l&15)
  int rsel = (lane >> 4) << 3;
  int colb = lane & 15;
#pragma unroll
  for (int j = 0; j < 8; ++j) {
    int rr = base + rsel + j;
    if (rr >= L) continue;
    long oi = out_idx[(long)k * L + rr];             // padded pairs scatter to dump row N
    float* d1 = acc1 + oi * DIMC + colb;
    float* d2 = acc2 + oi * DIMC + colb;
#pragma unroll
    for (int t = 0; t < 8; ++t) {
      atomicAdd(d1 + t * 16, c1[t][j]);
      atomicAdd(d2 + t * 16, c2[t][j]);
    }
  }
}

// ---------------- reductions: rm[n], bsum[b][c], bcnt[b] ----------------

__global__ __launch_bounds__(256)
void reduce_kernel(const float* __restrict__ acc1, const float* __restrict__ acc2,
                   const float* __restrict__ b1, const float* __restrict__ b2,
                   const int* __restrict__ batch_idx,
                   float* __restrict__ rm, float* __restrict__ bsum,
                   float* __restrict__ bcnt, int N) {
  __shared__ float sB[NBMAX * DIMC];
  __shared__ float sC[NBMAX];
  for (int i = threadIdx.x; i < NBMAX * DIMC; i += 256) sB[i] = 0.f;
  if (threadIdx.x < NBMAX) sC[threadIdx.x] = 0.f;
  __syncthreads();

  int wave = threadIdx.x >> 5, lane = threadIdx.x & 31;
  int row0 = blockIdx.x * 64;
  int rowEnd = (row0 + 64 < N) ? row0 + 64 : N;
  for (int rr = row0 + wave; rr < rowEnd; rr += 8) {
    int b = batch_idx[rr];
    float s = 0.f;
#pragma unroll
    for (int u = 0; u < 4; ++u) {
      int col = lane * 4 + u;
      long idx = (long)rr * DIMC + col;
      float v1 = fmaxf(acc1[idx] + b1[col], 0.f);
      float v2 = fmaxf(acc2[idx] + b2[col], 0.f);
      s += v1;
      atomicAdd(&sB[b * DIMC + col], v2);
    }
    for (int off = 16; off > 0; off >>= 1) s += __shfl_down(s, off, 32);
    if (lane == 0) { rm[rr] = s * (1.f / DIMC); atomicAdd(&sC[b], 1.f); }
  }
  __syncthreads();
  for (int i = threadIdx.x; i < NBMAX * DIMC; i += 256)
    if (sB[i] != 0.f) atomicAdd(&bsum[i], sB[i]);
  if (threadIdx.x < NBMAX && sC[threadIdx.x] != 0.f)
    atomicAdd(&bcnt[threadIdx.x], sC[threadIdx.x]);
}

// ---------------- f = sqrt(rm*bm) + out1 + out2 (bf16) ----------------

__global__ __launch_bounds__(256)
void make_f_kernel(const float* __restrict__ acc1, const float* __restrict__ acc2,
                   const float* __restrict__ b1, const float* __restrict__ b2,
                   const float* __restrict__ rm, const float* __restrict__ bsum,
                   const float* __restrict__ bcnt, const int* __restrict__ batch_idx,
                   bf16_t* __restrict__ f, int N) {
  long i = (long)blockIdx.x * 256 + threadIdx.x;
  if (i >= (long)N * DIMC) return;
  int n = (int)(i >> 7);
  int c = (int)(i & (DIMC - 1));
  float o1 = fmaxf(acc1[i] + b1[c], 0.f);
  float o2 = fmaxf(acc2[i] + b2[c], 0.f);
  int b = batch_idx[n];
  float bm = bsum[b * DIMC + c] / bcnt[b];
  float enc = sqrtf(rm[n] * bm + 1e-12f);
  f[i] = (bf16_t)(enc + o1 + o2);
}

// ---------------- out = relu(feats - relu(f @ W3 + b3)) ----------------

__global__ __launch_bounds__(256)
void out_gemm_kernel(const bf16_t* __restrict__ f, const bf16_t* __restrict__ pW3,
                     const float* __restrict__ b3, const float* __restrict__ feats,
                     float* __restrict__ out, int N) {
  __shared__ bf16_t sW3[16384];          // 32 KB
  {
    const f32x4* g = (const f32x4*)pW3;
    f32x4* s = (f32x4*)sW3;
    for (int i = threadIdx.x; i < 1024; i += 256) stage16(g + i, s + i);
  }
  stage_fence();

  int wave = threadIdx.x >> 5;
  int lane = threadIdx.x & 31;
  long gt = (long)blockIdx.x * 8 + wave;
  int base = (int)gt * 16;
  if (base >= N) return;                 // after the barrier: safe

  int na = base + (lane & 15);
  if (na >= N) na = N - 1;
  const bf16_t* arow = f + (long)na * DIMC;

  v8f c[8] = {};
#pragma unroll
  for (int cch = 0; cch < 4; ++cch) {
    v16bf a = load_a_frag(arow, cch, lane);
#pragma unroll
    for (int t = 0; t < 8; ++t) {
      v16bf bw = load_b_lds(sW3, cch, t, lane);
      c[t] = __builtin_amdgcn_wmma_f32_16x16x32_bf16(false, a, false, bw,
                                                     (short)0, c[t], false, false);
    }
  }

  int rsel = (lane >> 4) << 3;
  int colb = lane & 15;
#pragma unroll
  for (int j = 0; j < 8; ++j) {
    int n = base + rsel + j;
    if (n >= N) continue;
#pragma unroll
    for (int t = 0; t < 8; ++t) {
      int col = t * 16 + colb;
      float v = fmaxf(c[t][j] + b3[col], 0.f);
      out[(long)n * DIMC + col] = fmaxf(feats[(long)n * DIMC + col] - v, 0.f);
    }
  }
}

// ---------------- host launcher ----------------

extern "C" void kernel_launch(void* const* d_in, const int* in_sizes, int n_in,
                              void* d_out, int out_size, void* d_ws, size_t ws_size,
                              hipStream_t stream) {
  const float* feats     = (const float*)d_in[0];
  const float* W1        = (const float*)d_in[1];
  const float* b1        = (const float*)d_in[2];
  const float* W2        = (const float*)d_in[3];
  const float* b2        = (const float*)d_in[4];
  const float* W3        = (const float*)d_in[5];
  const float* b3        = (const float*)d_in[6];
  const int*   batch_idx = (const int*)d_in[7];
  const int*   in_idx    = (const int*)d_in[8];
  const int*   out_idx   = (const int*)d_in[9];

  const int N = in_sizes[0] / DIMC;
  const int L = in_sizes[8] / KOFF;

  // workspace carving (256B aligned)
  size_t off = 0;
  char* ws = (char*)d_ws;
  auto carve = [&](size_t bytes) -> char* {
    char* p = ws + off;
    off += (bytes + 255) & ~(size_t)255;
    return p;
  };
  bf16_t* featsb = (bf16_t*)carve((size_t)N * DIMC * sizeof(bf16_t)); // reused as f
  bf16_t* pW1    = (bf16_t*)carve((size_t)KOFF * 4 * 8 * 512 * sizeof(bf16_t));
  bf16_t* pW2    = (bf16_t*)carve((size_t)KOFF * 4 * 8 * 512 * sizeof(bf16_t));
  bf16_t* pW3    = (bf16_t*)carve((size_t)4 * 8 * 512 * sizeof(bf16_t));
  float*  acc1   = (float*)carve((size_t)(N + 1) * DIMC * sizeof(float));
  float*  acc2   = (float*)carve((size_t)(N + 1) * DIMC * sizeof(float));
  float*  rm     = (float*)carve((size_t)N * sizeof(float));
  float*  bsum   = (float*)carve((size_t)NBMAX * DIMC * sizeof(float));
  float*  bcnt   = (float*)carve((size_t)NBMAX * sizeof(float));

  // 1) precision prep: feats -> bf16, weights -> packed bf16 B-fragments
  long ntot = (long)N * DIMC;
  cast_bf16_kernel<<<(int)((ntot + 255) / 256), 256, 0, stream>>>(feats, featsb, ntot);
  long wtot = (long)KOFF * 4 * 8 * 512;
  pack_w_kernel<<<(int)((wtot + 255) / 256), 256, 0, stream>>>(W1, pW1, KOFF);
  pack_w_kernel<<<(int)((wtot + 255) / 256), 256, 0, stream>>>(W2, pW2, KOFF);
  pack_w_kernel<<<(16384 + 255) / 256, 256, 0, stream>>>(W3, pW3, 1);

  // 2) zero accumulators (graph-capturable async memsets)
  (void)hipMemsetAsync(acc1, 0, (size_t)(N + 1) * DIMC * sizeof(float), stream);
  (void)hipMemsetAsync(acc2, 0, (size_t)(N + 1) * DIMC * sizeof(float), stream);
  (void)hipMemsetAsync(bsum, 0, (size_t)NBMAX * DIMC * sizeof(float), stream);
  (void)hipMemsetAsync(bcnt, 0, (size_t)NBMAX * sizeof(float), stream);

  // 3) dual sparse conv (gather -> WMMA -> scatter-add), one k per block-row
  int tilesPerK = (L + 15) / 16;
  dim3 cgrid((tilesPerK + 7) / 8, KOFF);
  spconv_dual_kernel<<<cgrid, 256, 0, stream>>>(featsb, pW1, pW2, in_idx, out_idx,
                                                acc1, acc2, L, tilesPerK);

  // 4) reductions
  reduce_kernel<<<(N + 63) / 64, 256, 0, stream>>>(acc1, acc2, b1, b2, batch_idx,
                                                   rm, bsum, bcnt, N);

  // 5) fused feature (reuse featsb buffer as f, kernel 6 no longer needs featsb)
  make_f_kernel<<<(int)((ntot + 255) / 256), 256, 0, stream>>>(
      acc1, acc2, b1, b2, rm, bsum, bcnt, batch_idx, featsb, N);

  // 6) dense 1x1 conv + residual epilogue
  int otiles = (N + 15) / 16;
  out_gemm_kernel<<<(otiles + 7) / 8, 256, 0, stream>>>(featsb, pW3, b3, feats,
                                                        (float*)d_out, N);
  (void)n_in; (void)out_size; (void)ws_size; (void)batch_idx;
}
